// BotSpot_72816875537006
// MI455X (gfx1250) — matrix operand order repaired and checked by reading, not compiled, not run on hardware
//
#include <hip/hip_runtime.h>

typedef __attribute__((ext_vector_type(16))) _Float16 v16h;
typedef __attribute__((ext_vector_type(8)))  _Float16 v8h;
typedef __attribute__((ext_vector_type(8)))  float    v8f;
typedef __attribute__((ext_vector_type(4)))  float    v4f;

#define NBATCH   8192
#define NBR      100
#define D_DEV    113
#define D_MSGD   67
#define K_PAD    128
#define N_PAD    80
#define M_PAD    112

// phase-2 dims
#define D_COMBX  46
#define D_CH     27
#define D_FUS    56
#define D_DEV1   67
#define D_DEV2   50
#define CAT_IN   106
#define D_C1     63
#define D_C2     31

// workspace layout
#define WS_WPT_OFF   0          // 80*128 f16 = 20480 B
#define WS_MSG_OFF   32768      // 8192*80 f32 = 2.62 MB

// K-permutation: k' in [0,112) -> feature k'+1 (table t = k'>>4, elem j = k'&15)
//                k' == 112     -> feature 0 (continuous)
//                k' in (112,128) -> zero pad
// Applied identically to the A tile and to W_msg, so the dot product is unchanged.

__device__ __forceinline__ const float* tab_sel(int t,
    const float* t0, const float* t1, const float* t2, const float* t3,
    const float* t4, const float* t5, const float* t6) {
    switch (t) { case 0: return t0; case 1: return t1; case 2: return t2;
                 case 3: return t3; case 4: return t4; case 5: return t5;
                 default: return t6; }
}

// ---------------------------------------------------------------------------
// Kernel 0: W_msg (67x113 f32) -> permuted/padded WpT[80][128] f16
// ---------------------------------------------------------------------------
__global__ void prep_wmsg(const float* __restrict__ W_msg, _Float16* __restrict__ WpT) {
    int i = blockIdx.x * blockDim.x + threadIdx.x;
    if (i >= N_PAD * K_PAD) return;
    int n = i >> 7, kp = i & (K_PAD - 1);
    float v = 0.0f;
    if (n < D_MSGD) {
        if (kp < 112)       v = W_msg[n * D_DEV + kp + 1];
        else if (kp == 112) v = W_msg[n * D_DEV + 0];
    }
    WpT[i] = (_Float16)v;
}

// ---------------------------------------------------------------------------
// Kernel 1: neighbor message GEMM + ReLU + mean, one block per edge.
// ---------------------------------------------------------------------------
__global__ __launch_bounds__(128)
void msg_kernel(const float* __restrict__ device_feats,
                const float* __restrict__ t0, const float* __restrict__ t1,
                const float* __restrict__ t2, const float* __restrict__ t3,
                const float* __restrict__ t4, const float* __restrict__ t5,
                const float* __restrict__ t6,
                const _Float16* __restrict__ WpT,
                const float* __restrict__ b_msg,
                const int* __restrict__ neibrs,
                float* __restrict__ msg_out)
{
    __shared__ __align__(16) _Float16 sA[M_PAD][K_PAD];   // 28672 B
    __shared__ float sSum[N_PAD];

    const int b   = blockIdx.x;
    const int tid = threadIdx.x;

    if (tid < N_PAD) sSum[tid] = 0.0f;

    // ---- build A tile: 8 threads per row, 16 rows per pass, 7 passes ----
    // thread sub 0..6: embedding table `sub` -> k' = [16*sub, 16*sub+16), vectorized
    // thread sub 7   : continuous feature at k'=112 + zero pad to 128
    const int sub   = tid & 7;
    const int rbase = tid >> 3;     // 0..15
    const v8h zero8 = {};
    for (int pass = 0; pass < 7; ++pass) {
        const int r = pass * 16 + rbase;       // 0..111
        _Float16* dst = &sA[r][sub * 16];
        if (r < NBR) {
            const int    nrow = neibrs[b * NBR + r];
            const float* df   = device_feats + (long)nrow * 8;
            if (sub < 7) {
                const int c = (int)df[1 + sub];
                const float* trow = tab_sel(sub, t0, t1, t2, t3, t4, t5, t6) + c * 16;
                const v4f f0 = *(const v4f*)(trow + 0);
                const v4f f1 = *(const v4f*)(trow + 4);
                const v4f f2 = *(const v4f*)(trow + 8);
                const v4f f3 = *(const v4f*)(trow + 12);
                v8h lo, hi;
                #pragma unroll
                for (int i = 0; i < 4; ++i) {
                    lo[i]     = (_Float16)f0[i];
                    lo[4 + i] = (_Float16)f1[i];
                    hi[i]     = (_Float16)f2[i];
                    hi[4 + i] = (_Float16)f3[i];
                }
                *(v8h*)(dst)     = lo;
                *(v8h*)(dst + 8) = hi;
            } else {
                v8h lo = {};
                lo[0] = (_Float16)df[0];        // cont feature at k'=112
                *(v8h*)(dst)     = lo;
                *(v8h*)(dst + 8) = zero8;
            }
        } else {
            *(v8h*)(dst)     = zero8;
            *(v8h*)(dst + 8) = zero8;
        }
    }
    __syncthreads();

    // ---- WMMA over 7x5 MN tiles, split across 4 waves ----
    const int wave  = tid >> 5;
    const int lane  = tid & 31;
    const int lhalf = (lane < 16) ? 0 : 1;
    const int l15   = lane & 15;

    for (int tile = wave; tile < 35; tile += 4) {
        const int mt = tile / 5;
        const int nt = tile % 5;
        const int arow = mt * 16 + l15;
        const int ncol = nt * 16 + l15;

        v8f acc = {};
        #pragma unroll
        for (int kc = 0; kc < 4; ++kc) {
            // A fragment (16-bit A layout): lanes 0-15 hold K = kc*32 + {0..7, 16..23},
            // lanes 16-31 hold K = kc*32 + {8..15, 24..31}
            const int ka = kc * 32 + lhalf * 8;
            union { v16h v; v8h h[2]; } ua;
            ua.h[0] = *(const v8h*)&sA[arow][ka];
            ua.h[1] = *(const v8h*)&sA[arow][ka + 16];

            // B fragment: same K split as A ({0..7,16..23} / {8..15,24..31})
            const int kb = kc * 32 + lhalf * 8;
            union { v16h v; v8h h[2]; } ub;
            ub.h[0] = *(const v8h*)(WpT + ncol * K_PAD + kb);
            ub.h[1] = *(const v8h*)(WpT + ncol * K_PAD + kb + 16);
            v16h bb = ub.v;

            acc = __builtin_amdgcn_wmma_f32_16x16x32_f16(
                      false, ua.v, false, bb, (short)0, acc, false, false);
        }

        // C/D layout: VGPR r -> row mt*16 + r (lanes 0-15) / +8 (lanes 16-31)
        if (ncol < D_MSGD) {
            const float bn = b_msg[ncol];
            float s = 0.0f;
            #pragma unroll
            for (int r = 0; r < 8; ++r) {
                const int row = mt * 16 + r + lhalf * 8;
                if (row < NBR) {
                    const float v = acc[r] + bn;
                    s += (v > 0.0f) ? v : 0.0f;
                }
            }
            atomicAdd(&sSum[ncol], s);
        }
    }
    __syncthreads();

    if (tid < D_MSGD) msg_out[b * N_PAD + tid] = sSum[tid] * (1.0f / NBR);
}

// ---------------------------------------------------------------------------
// Kernel 2: per-edge MLP chain, one block per edge (128 threads).
// ---------------------------------------------------------------------------
__device__ __forceinline__ void layer_dot(const float* __restrict__ W,
                                          const float* __restrict__ bias,
                                          const float* in, int din,
                                          float* out, int dout, int tid)
{
    for (int o = tid; o < dout; o += 128) {
        const float* w = W + o * din;
        float s = bias[o];
        for (int i = 0; i < din; ++i) s += w[i] * in[i];
        out[o] = (s > 0.0f) ? s : 0.0f;
    }
}

__global__ __launch_bounds__(128)
void mlp_kernel(const float* __restrict__ combin_feats,
                const float* __restrict__ device_feats,
                const float* __restrict__ ch_emb,
                const float* __restrict__ t0, const float* __restrict__ t1,
                const float* __restrict__ t2, const float* __restrict__ t3,
                const float* __restrict__ t4, const float* __restrict__ t5,
                const float* __restrict__ t6,
                const float* __restrict__ W_ch1, const float* __restrict__ b_ch1,
                const float* __restrict__ W_fus, const float* __restrict__ b_fus,
                const float* __restrict__ W_dev1, const float* __restrict__ b_dev1,
                const float* __restrict__ W_dev2, const float* __restrict__ b_dev2,
                const float* __restrict__ W_c1, const float* __restrict__ b_c1,
                const float* __restrict__ W_c2, const float* __restrict__ b_c2,
                const float* __restrict__ W_c3, const float* __restrict__ b_c3,
                const int* __restrict__ edges,
                const float* __restrict__ msg_in,
                float* __restrict__ out)
{
    __shared__ float xc[D_COMBX], xd[D_DEV], xm[D_MSGD];
    __shared__ float ch[D_CH], fus[D_FUS], d1[D_DEV1], d2[D_DEV2];
    __shared__ float h1[D_C1], h2[D_C2];

    const int b   = blockIdx.x;
    const int tid = threadIdx.x;
    const int cidx = edges[2 * b];
    const int didx = edges[2 * b + 1];

    // combin_x = [30 cont feats, channel_id_emb[id]]
    if (tid < 30) xc[tid] = combin_feats[(long)cidx * 31 + tid];
    if (tid >= 30 && tid < D_COMBX) {
        const int chid = (int)combin_feats[(long)cidx * 31 + 30];
        xc[tid] = ch_emb[chid * 16 + (tid - 30)];
    }
    // dev_x = [cont, 7 x 16-wide embeddings]  (reference feature order)
    if (tid < D_DEV) {
        float v;
        if (tid == 0) {
            v = device_feats[(long)didx * 8];
        } else {
            const int t = (tid - 1) >> 4;
            const int j = (tid - 1) & 15;
            const int c = (int)device_feats[(long)didx * 8 + 1 + t];
            v = tab_sel(t, t0, t1, t2, t3, t4, t5, t6)[c * 16 + j];
        }
        xd[tid] = v;
    }
    if (tid < D_MSGD) xm[tid] = msg_in[b * N_PAD + tid];
    __syncthreads();

    layer_dot(W_ch1, b_ch1, xc, D_COMBX, ch, D_CH, tid);
    layer_dot(W_dev1, b_dev1, xd, D_DEV, d1, D_DEV1, tid);
    __syncthreads();

    // fus = relu(W_fus @ [ch; msg] + b)
    for (int o = tid; o < D_FUS; o += 128) {
        const float* w = W_fus + o * (D_CH + D_MSGD);
        float s = b_fus[o];
        for (int i = 0; i < D_CH; ++i)   s += w[i] * ch[i];
        for (int i = 0; i < D_MSGD; ++i) s += w[D_CH + i] * xm[i];
        fus[o] = (s > 0.0f) ? s : 0.0f;
    }
    layer_dot(W_dev2, b_dev2, d1, D_DEV1, d2, D_DEV2, tid);
    __syncthreads();

    // h1 = relu(W_c1 @ [fus; d2] + b)
    for (int o = tid; o < D_C1; o += 128) {
        const float* w = W_c1 + o * CAT_IN;
        float s = b_c1[o];
        for (int i = 0; i < D_FUS; ++i)  s += w[i] * fus[i];
        for (int i = 0; i < D_DEV2; ++i) s += w[D_FUS + i] * d2[i];
        h1[o] = (s > 0.0f) ? s : 0.0f;
    }
    __syncthreads();

    layer_dot(W_c2, b_c2, h1, D_C1, h2, D_C2, tid);
    __syncthreads();

    if (tid == 0) {
        float s = b_c3[0];
        for (int i = 0; i < D_C2; ++i) s += W_c3[i] * h2[i];
        out[b] = s;
    }
}

// ---------------------------------------------------------------------------
extern "C" void kernel_launch(void* const* d_in, const int* in_sizes, int n_in,
                              void* d_out, int out_size, void* d_ws, size_t ws_size,
                              hipStream_t stream) {
    const float* combin_feats = (const float*)d_in[0];
    const float* device_feats = (const float*)d_in[1];
    const float* ch_emb       = (const float*)d_in[2];
    const float* t0 = (const float*)d_in[3];   // lang
    const float* t1 = (const float*)d_in[4];   // plat
    const float* t2 = (const float*)d_in[5];   // os
    const float* t3 = (const float*)d_in[6];   // country
    const float* t4 = (const float*)d_in[7];   // carrier
    const float* t5 = (const float*)d_in[8];   // brand
    const float* t6 = (const float*)d_in[9];   // plat_os
    const float* W_ch1  = (const float*)d_in[10];
    const float* b_ch1  = (const float*)d_in[11];
    const float* W_msg  = (const float*)d_in[12];
    const float* b_msg  = (const float*)d_in[13];
    const float* W_fus  = (const float*)d_in[14];
    const float* b_fus  = (const float*)d_in[15];
    const float* W_dev1 = (const float*)d_in[16];
    const float* b_dev1 = (const float*)d_in[17];
    const float* W_dev2 = (const float*)d_in[18];
    const float* b_dev2 = (const float*)d_in[19];
    const float* W_c1   = (const float*)d_in[20];
    const float* b_c1   = (const float*)d_in[21];
    const float* W_c2   = (const float*)d_in[22];
    const float* b_c2   = (const float*)d_in[23];
    const float* W_c3   = (const float*)d_in[24];
    const float* b_c3   = (const float*)d_in[25];
    const int*   edges  = (const int*)d_in[26];
    const int*   neibrs = (const int*)d_in[27];

    _Float16* WpT    = (_Float16*)((char*)d_ws + WS_WPT_OFF);
    float*    msg_ws = (float*)((char*)d_ws + WS_MSG_OFF);
    float*    outp   = (float*)d_out;

    prep_wmsg<<<(N_PAD * K_PAD + 255) / 256, 256, 0, stream>>>(W_msg, WpT);

    msg_kernel<<<NBATCH, 128, 0, stream>>>(device_feats,
        t0, t1, t2, t3, t4, t5, t6, WpT, b_msg, neibrs, msg_ws);

    mlp_kernel<<<NBATCH, 128, 0, stream>>>(combin_feats, device_feats, ch_emb,
        t0, t1, t2, t3, t4, t5, t6,
        W_ch1, b_ch1, W_fus, b_fus, W_dev1, b_dev1, W_dev2, b_dev2,
        W_c1, b_c1, W_c2, b_c2, W_c3, b_c3,
        edges, msg_ws, outp);
}
